// Attention_52690658787451
// MI455X (gfx1250) — compile-verified, matrix operand order checked by reading, not run on hardware
//
#include <hip/hip_runtime.h>
#include <math.h>

// Problem constants
#define BB 64
#define SS 2048
#define DD 512
#define AA 512

typedef __attribute__((ext_vector_type(16))) __bf16          v16bf;
typedef __attribute__((ext_vector_type(16))) unsigned short  v16us;
typedef __attribute__((ext_vector_type(8)))  unsigned int    v8u;
typedef __attribute__((ext_vector_type(8)))  float           v8f;

__device__ __forceinline__ unsigned short f2bf(float f) {
    unsigned u = __builtin_bit_cast(unsigned, f);
    return (unsigned short)((u + 0x7FFFu + ((u >> 16) & 1u)) >> 16);
}

__device__ __forceinline__ unsigned pack2bf(float a, float b) {
    return (unsigned)f2bf(a) | ((unsigned)f2bf(b) << 16);
}

__device__ __forceinline__ float fast_tanh(float x) {
    // tanh via single v_exp_f32 (TRANS co-executes with 16-bit WMMA)
    float e = __expf(-2.0f * fabsf(x));
    float t = (1.0f - e) / (1.0f + e);
    return copysignf(t, x);
}

// ---------------------------------------------------------------------------
// Kernel 0: pack Ws [D,A] f32 -> bf16 in WMMA B-fragment order.
// Layout: idx = ((atile*16 + kc)*32 + lane)*16 + e
//   lane: n = lane&15, g = lane>>4
//   e:    K = kc*32 + (e<8?0:16) + g*8 + (e&7), a = atile*16 + n
// ---------------------------------------------------------------------------
__global__ void pack_ws_kernel(const float* __restrict__ Ws,
                               unsigned short* __restrict__ wsbf) {
    int t    = blockIdx.x;          // 512 = 32 atiles * 16 kc
    int lane = threadIdx.x;         // 32
    int atile = t >> 4;
    int kc    = t & 15;
    int n = lane & 15;
    int g = lane >> 4;
    int a = atile * 16 + n;
    unsigned short* out = wsbf + ((size_t)(t * 32 + lane) << 4);
#pragma unroll
    for (int e = 0; e < 16; ++e) {
        int k = kc * 32 + ((e < 8) ? 0 : 16) + g * 8 + (e & 7);
        out[e] = f2bf(Ws[(size_t)k * AA + a]);
    }
}

// ---------------------------------------------------------------------------
// Kernel 1: hpb[b][a] = input[b,:] @ Wh[:,a] + b_attn[a]
// ---------------------------------------------------------------------------
__global__ void hproj_kernel(const float* __restrict__ input,
                             const float* __restrict__ Wh,
                             const float* __restrict__ b_attn,
                             float* __restrict__ hpb) {
    int b = blockIdx.x;
    int a = blockIdx.y * 256 + threadIdx.x;
    float s = b_attn[a];
    const float* in = input + (size_t)b * DD;
#pragma unroll 4
    for (int d = 0; d < DD; ++d)
        s = fmaf(in[d], Wh[(size_t)d * AA + a], s);
    hpb[(size_t)b * AA + a] = s;
}

// ---------------------------------------------------------------------------
// Kernel 2: main score kernel.
// Each wave32 owns 16 s-rows of one batch. A = context tile (16 x 512) bf16:
// k-chunks 0..7 resident in VGPRs, k-chunks 8..15 staged in LDS (8KB/wave,
// wave-private, bank-swizzled). B = packed Ws fragments from L2/WGP$.
// scores[b][s] = va . tanh(ctx@Ws + hpb[b]) with zero-mean mask -> -inf.
// ---------------------------------------------------------------------------
__global__ void __launch_bounds__(256)
attn_score_kernel(const float* __restrict__ context,
                  const unsigned short* __restrict__ wsbf,
                  const float* __restrict__ hpb,
                  const float* __restrict__ va,
                  float* __restrict__ scores) {
    extern __shared__ char smem[];                 // 64 KB: 8 waves * 8 KB

    int lane = threadIdx.x & 31;
    int wid  = threadIdx.x >> 5;                   // 8 waves / block
    int b    = blockIdx.x >> 4;                    // 64 batches
    int sblk = blockIdx.x & 15;                    // 16 s-blocks of 128
    int sbase = sblk * 128 + wid * 16;

    int n = lane & 15;
    int g = lane >> 4;

    // Bank-swizzled wave-private LDS chunk offsets (bijective per lane/half)
    char* abase = smem + wid * 8192;
    int c0 = ((2 * lane + 0) ^ ((lane >> 2) << 1)) * 16;
    int c1 = ((2 * lane + 1) ^ ((lane >> 2) << 1)) * 16;

    // ---- Load A fragments: row m = n, K-halves selected by g ----
    const float* rowptr = context + ((size_t)b * SS + sbase + n) * DD;
    v16bf afr[8];                                  // k-chunks 0..7 in VGPRs
    float lanesum = 0.0f;
#pragma unroll
    for (int kc = 0; kc < 16; ++kc) {
        const float* p0 = rowptr + kc * 32 + g * 8;    // K = kc*32 + 8g + [0..7]
        const float* p1 = p0 + 16;                     // K = kc*32 + 16 + 8g + [0..7]
        float4 x0 = *(const float4*)(p0);
        float4 x1 = *(const float4*)(p0 + 4);
        float4 y0 = *(const float4*)(p1);
        float4 y1 = *(const float4*)(p1 + 4);
        lanesum += x0.x + x0.y + x0.z + x0.w + x1.x + x1.y + x1.z + x1.w
                 + y0.x + y0.y + y0.z + y0.w + y1.x + y1.y + y1.z + y1.w;
        v8u au;
        au[0] = pack2bf(x0.x, x0.y);  au[1] = pack2bf(x0.z, x0.w);
        au[2] = pack2bf(x1.x, x1.y);  au[3] = pack2bf(x1.z, x1.w);
        au[4] = pack2bf(y0.x, y0.y);  au[5] = pack2bf(y0.z, y0.w);
        au[6] = pack2bf(y1.x, y1.y);  au[7] = pack2bf(y1.z, y1.w);
        if (kc < 8) {
            afr[kc] = __builtin_bit_cast(v16bf, au);
        } else {
            char* dst = abase + (kc - 8) * 1024;
            ((uint4*)(dst + c0))[0] = make_uint4(au[0], au[1], au[2], au[3]);
            ((uint4*)(dst + c1))[0] = make_uint4(au[4], au[5], au[6], au[7]);
        }
    }
    // rowsum[m]: lane m and lane m+16 hold complementary K halves
    float rowsum = lanesum + __shfl_xor(lanesum, 16);

    // ---- GEMM over 32 a-tiles, fused tanh . va epilogue ----
    float sum[8];
#pragma unroll
    for (int r = 0; r < 8; ++r) sum[r] = 0.0f;

    const v16us* bbase = (const v16us*)wsbf;   // one v16us = one lane B-slice
#pragma unroll 1
    for (int atile = 0; atile < 32; ++atile) {
        const v16us* bp = bbase + ((size_t)(atile * 16) * 32 + lane);
        v8f acc = {};
        // k-chunks 0..7: A from registers
#pragma unroll
        for (int kc = 0; kc < 8; ++kc) {
            v16bf bfr = __builtin_bit_cast(v16bf, bp[kc * 32]);
            acc = __builtin_amdgcn_wmma_f32_16x16x32_bf16(
                      false, afr[kc], false, bfr, (short)0, acc, false, false);
        }
        // k-chunks 8..15: A from LDS (wave-private, two ds_load_b128 each)
#pragma unroll
        for (int kc = 8; kc < 16; ++kc) {
            v16bf bfr = __builtin_bit_cast(v16bf, bp[kc * 32]);
            const char* src = abase + (kc - 8) * 1024;
            uint4 lo = ((const uint4*)(src + c0))[0];
            uint4 hi = ((const uint4*)(src + c1))[0];
            v8u au;
            au[0] = lo.x; au[1] = lo.y; au[2] = lo.z; au[3] = lo.w;
            au[4] = hi.x; au[5] = hi.y; au[6] = hi.z; au[7] = hi.w;
            v16bf afl = __builtin_bit_cast(v16bf, au);
            acc = __builtin_amdgcn_wmma_f32_16x16x32_bf16(
                      false, afl, false, bfr, (short)0, acc, false, false);
        }
        int a = atile * 16 + n;
        float vav = va[a];
        float hb  = hpb[(size_t)b * AA + a];
#pragma unroll
        for (int r = 0; r < 8; ++r)
            sum[r] = fmaf(fast_tanh(acc[r] + hb), vav, sum[r]);
    }

    // ---- Reduce over a-columns (n = 0..15 within each 16-lane half) ----
#pragma unroll
    for (int r = 0; r < 8; ++r) {
        sum[r] += __shfl_xor(sum[r], 1);
        sum[r] += __shfl_xor(sum[r], 2);
        sum[r] += __shfl_xor(sum[r], 4);
        sum[r] += __shfl_xor(sum[r], 8);
    }

    // ---- Apply padding mask, write scores (all lanes shuffle, n==0 stores) ----
    float outv[8];
#pragma unroll
    for (int r = 0; r < 8; ++r) {
        int m = r + 8 * g;                 // rows 0..7 (g=0) / 8..15 (g=1)
        float rs = __shfl(rowsum, m);      // lane m holds rowsum for row m
        outv[r] = (rs == 0.0f) ? -__builtin_inff() : sum[r];
    }
    if (n == 0) {
#pragma unroll
        for (int r = 0; r < 8; ++r)
            scores[(size_t)b * SS + sbase + r + 8 * g] = outv[r];
    }
}

// ---------------------------------------------------------------------------
// Kernel 3: in-place softmax over S per batch row.
// ---------------------------------------------------------------------------
__global__ void softmax_kernel(float* __restrict__ dist) {
    __shared__ float red[256];
    int b = blockIdx.x;
    int tid = threadIdx.x;
    float* row = dist + (size_t)b * SS;

    float mx = -__builtin_inff();
    for (int s = tid; s < SS; s += 256) mx = fmaxf(mx, row[s]);
    red[tid] = mx; __syncthreads();
    for (int off = 128; off > 0; off >>= 1) {
        if (tid < off) red[tid] = fmaxf(red[tid], red[tid + off]);
        __syncthreads();
    }
    mx = red[0]; __syncthreads();

    float sm = 0.0f;
    for (int s = tid; s < SS; s += 256) sm += __expf(row[s] - mx);
    red[tid] = sm; __syncthreads();
    for (int off = 128; off > 0; off >>= 1) {
        if (tid < off) red[tid] += red[tid + off];
        __syncthreads();
    }
    float inv = 1.0f / red[0];
    for (int s = tid; s < SS; s += 256) row[s] = __expf(row[s] - mx) * inv;
}

// ---------------------------------------------------------------------------
// Kernel 4: partial weighted context sums: part[b][ch][d] over 128-s chunks.
// ---------------------------------------------------------------------------
__global__ void ctx_partial_kernel(const float* __restrict__ context,
                                   const float* __restrict__ dist,
                                   float* __restrict__ part) {
    int b  = blockIdx.x;
    int ch = blockIdx.y;       // 16 chunks of 128
    int d0 = threadIdx.x;      // 256 threads, 2 columns each
    float a0 = 0.0f, a1 = 0.0f;
    int s0 = ch * 128;
    for (int s = s0; s < s0 + 128; ++s) {
        float w = dist[(size_t)b * SS + s];
        const float* cp = context + ((size_t)b * SS + s) * DD;
        a0 = fmaf(w, cp[d0],        a0);
        a1 = fmaf(w, cp[d0 + 256],  a1);
    }
    float* pp = part + ((size_t)(b * 16 + ch)) * DD;
    pp[d0]       = a0;
    pp[d0 + 256] = a1;
}

// ---------------------------------------------------------------------------
// Kernel 5: fixed-order reduction of the 16 partials -> attn_context.
// ---------------------------------------------------------------------------
__global__ void ctx_reduce_kernel(const float* __restrict__ part,
                                  float* __restrict__ out) {
    int b = blockIdx.x;
    int tid = threadIdx.x;
#pragma unroll
    for (int h = 0; h < 2; ++h) {
        int d = tid + h * 256;
        float s = 0.0f;
#pragma unroll
        for (int ch = 0; ch < 16; ++ch)
            s += part[((size_t)(b * 16 + ch)) * DD + d];
        out[(size_t)b * DD + d] = s;
    }
}

// ---------------------------------------------------------------------------
extern "C" void kernel_launch(void* const* d_in, const int* in_sizes, int n_in,
                              void* d_out, int out_size, void* d_ws, size_t ws_size,
                              hipStream_t stream) {
    const float* input   = (const float*)d_in[0];   // [B, D]
    const float* context = (const float*)d_in[1];   // [B, S, D]
    const float* Wh      = (const float*)d_in[2];   // [D, A]
    const float* Ws      = (const float*)d_in[3];   // [D, A]
    const float* b_attn  = (const float*)d_in[4];   // [A]
    const float* va      = (const float*)d_in[5];   // [A]
    float* out = (float*)d_out;                     // [B*D ctx | B*S dist]

    // Workspace layout (bytes)
    char* ws = (char*)d_ws;
    float*          hpb  = (float*)ws;                               // B*A f32   = 128 KB
    unsigned short* wsbf = (unsigned short*)(ws + (size_t)BB*AA*4);  // D*A bf16  = 512 KB
    float*          part = (float*)(ws + (size_t)BB*AA*4
                                       + (size_t)DD*AA*2);           // B*16*D f32 = 2 MB

    float* attn_ctx  = out;                 // [B, D]
    float* attn_dist = out + (size_t)BB*DD; // [B, S] (scores in-place -> softmax)

    pack_ws_kernel<<<512, 32, 0, stream>>>(Ws, wsbf);
    hproj_kernel<<<dim3(BB, AA / 256), 256, 0, stream>>>(input, Wh, b_attn, hpb);
    attn_score_kernel<<<BB * (SS / 128), 256, 64 * 1024, stream>>>(context, wsbf, hpb, va, attn_dist);
    softmax_kernel<<<BB, 256, 0, stream>>>(attn_dist);
    ctx_partial_kernel<<<dim3(BB, 16), 256, 0, stream>>>(context, attn_dist, part);
    ctx_reduce_kernel<<<BB, 256, 0, stream>>>(part, attn_ctx);
}
